// MambaPatchExpand_41618233098416
// MI455X (gfx1250) — matrix operand order repaired and checked by reading, not compiled, not run on hardware
//
#include <hip/hip_runtime.h>
#include <hip/hip_bf16.h>

// MambaPatchExpand for MI455X (gfx1250): WMMA f16 GEMMs (guard-free, float4
// fragment loads, pre-transposed weights) + register-resident selective scan.

#define BB 8
#define HH 64
#define WW 64
#define CC 96
#define DD 192
#define NSTATE 16
#define RRANK 6
#define KDIRS 4
#define LL (HH * WW)       // 4096
#define MTOT (BB * LL)     // 32768
#define XDBL_W 48          // dt_rank + 2*N = 38, padded to 48 (3 N-tiles of 16)

typedef _Float16 v16h __attribute__((ext_vector_type(16)));
typedef float    v8f  __attribute__((ext_vector_type(8)));

// ---- workspace layout (float offsets) ----
#define OFF_XZ    0L
#define SZ_XZ     ((long)MTOT * 2 * DD)           // 12,582,912  xz [b*p,384]; reused for xp
#define OFF_XC    (OFF_XZ + SZ_XZ)
#define SZ_XC     ((long)MTOT * DD)               // conv output [b*p,192]
#define OFF_XDBL  (OFF_XC + SZ_XC)
#define SZ_XDBL   ((long)KDIRS * MTOT * XDBL_W)   // [k][b*p][48]
#define OFF_YK    (OFF_XDBL + SZ_XDBL)
#define SZ_YK     ((long)MTOT * DD)               // x4 dirs; yk0 reused as yg, yk1 as out1
#define OFF_WT    (OFF_YK + 4 * SZ_YK)
#define SZ_WIN_T  ((long)(2 * DD) * CC)           // W_in^T  [384,96]
#define SZ_WOUT_T ((long)CC * DD)                 // W_out^T [96,192]
#define SZ_WEXP_T ((long)(2 * CC) * CC)           // W_exp^T [192,96]
#define SZ_XPW_P  ((long)KDIRS * XDBL_W * DD)     // padded x_proj_weight [4,48,192]
// total ~= 50.4M floats ~= 192 MB

__device__ __forceinline__ float wred(float v) {
#pragma unroll
  for (int off = 16; off > 0; off >>= 1) v += __shfl_xor(v, off, 32);
  return v;
}

__device__ __forceinline__ float silu_f(float x) { return x / (1.f + __expf(-x)); }

// ---------------------------------------------------------------------------
// Weight prep: generic transpose (dst[c,r] = src[r,c]) and x_proj pad.
// ---------------------------------------------------------------------------
__global__ void transpose_kernel(const float* __restrict__ src, float* __restrict__ dst,
                                 int rows, int cols) {
  long i = (long)blockIdx.x * blockDim.x + threadIdx.x;
  if (i >= (long)rows * cols) return;
  int c = (int)(i % cols);
  int r = (int)(i / cols);
  dst[(long)c * rows + r] = src[i];
}

__global__ void pad_xpw_kernel(const float* __restrict__ src, float* __restrict__ dst) {
  // src [4,38,192] -> dst [4,48,192]; rows 38..47 zero-filled
  long i = (long)blockIdx.x * blockDim.x + threadIdx.x;
  if (i >= (long)KDIRS * XDBL_W * DD) return;
  int col = (int)(i % DD);
  long t = i / DD;
  int row = (int)(t % XDBL_W);
  int k = (int)(t / XDBL_W);
  dst[i] = (row < 38) ? src[((long)k * 38 + row) * DD + col] : 0.f;
}

// ---------------------------------------------------------------------------
// WMMA GEMM: C[M,N] = A[M,K] * Bt[N,K]^T, fp32 memory, f16 compute, f32 acc.
// Requirements (all satisfied by call sites): K % 32 == 0, M % 16 == 0,
// N % 16 == 0 (x_proj padded), lda = ldb = K. No guards -> straight-line
// float4 loads feeding v_wmma_f32_16x16x32_f16; K loop fully unrolled.
// A fragment (ISA 7.12.2, 16-bit A 16x32): lane half-group g gets the two
// 8-consecutive-K runs [k0+8g, +8) and [k0+16+8g, +8).
// B fragment: lane owns column nc, same K runs as A: [k0+8g, +8) and [k0+16+8g, +8).
// ---------------------------------------------------------------------------
template <int KDIM>
__global__ __launch_bounds__(128) void wmma_gemm_kernel(
    const float* __restrict__ A, const float* __restrict__ Bt,
    float* __restrict__ Cmat, int tilesM, int tilesN, int ldc,
    long strideB, long strideC) {
  int tile = blockIdx.x * (blockDim.x >> 5) + (threadIdx.x >> 5);
  if (tile >= tilesM * tilesN) return;  // wave-uniform: EXEC all-1s at WMMA
  const float* Bp = Bt + (long)blockIdx.y * strideB;
  float* Cp = Cmat + (long)blockIdx.y * strideC;
  int tm = tile / tilesN, tn = tile % tilesN;
  int lane = threadIdx.x & 31;
  int g = lane >> 4;
  int mr = tm * 16 + (lane & 15);
  int nc = tn * 16 + (lane & 15);
  const float* Arow = A + (long)mr * KDIM;
  const float* Brow = Bp + (long)nc * KDIM;
  v8f acc = {};
#pragma unroll
  for (int k0 = 0; k0 < KDIM; k0 += 32) {
    const float4 a0 = *(const float4*)(Arow + k0 + 8 * g);
    const float4 a1 = *(const float4*)(Arow + k0 + 8 * g + 4);
    const float4 a2 = *(const float4*)(Arow + k0 + 16 + 8 * g);
    const float4 a3 = *(const float4*)(Arow + k0 + 16 + 8 * g + 4);
    const float4 b0 = *(const float4*)(Brow + k0 + 8 * g);
    const float4 b1 = *(const float4*)(Brow + k0 + 8 * g + 4);
    const float4 b2 = *(const float4*)(Brow + k0 + 16 + 8 * g);
    const float4 b3 = *(const float4*)(Brow + k0 + 16 + 8 * g + 4);
    v16h af, bf;
    af[0] = (_Float16)a0.x; af[1] = (_Float16)a0.y; af[2] = (_Float16)a0.z; af[3] = (_Float16)a0.w;
    af[4] = (_Float16)a1.x; af[5] = (_Float16)a1.y; af[6] = (_Float16)a1.z; af[7] = (_Float16)a1.w;
    af[8] = (_Float16)a2.x; af[9] = (_Float16)a2.y; af[10] = (_Float16)a2.z; af[11] = (_Float16)a2.w;
    af[12] = (_Float16)a3.x; af[13] = (_Float16)a3.y; af[14] = (_Float16)a3.z; af[15] = (_Float16)a3.w;
    bf[0] = (_Float16)b0.x; bf[1] = (_Float16)b0.y; bf[2] = (_Float16)b0.z; bf[3] = (_Float16)b0.w;
    bf[4] = (_Float16)b1.x; bf[5] = (_Float16)b1.y; bf[6] = (_Float16)b1.z; bf[7] = (_Float16)b1.w;
    bf[8] = (_Float16)b2.x; bf[9] = (_Float16)b2.y; bf[10] = (_Float16)b2.z; bf[11] = (_Float16)b2.w;
    bf[12] = (_Float16)b3.x; bf[13] = (_Float16)b3.y; bf[14] = (_Float16)b3.z; bf[15] = (_Float16)b3.w;
    acc = __builtin_amdgcn_wmma_f32_16x16x32_f16(
        /*neg_a=*/false, af, /*neg_b=*/false, bf,
        /*c_mod=*/(short)0, acc, /*reuse_a=*/false, /*reuse_b=*/false);
  }
#pragma unroll
  for (int r = 0; r < 8; r++)  // C/D: VGPR r -> row r + 8g, col = lane&15
    Cp[(long)(tm * 16 + r + 8 * g) * ldc + nc] = acc[r];
}

// ---------------------------------------------------------------------------
// Depthwise 3x3 conv (SAME) + SiLU. Reads xc half of xz [b*p,384] -> [b*p,192].
// ---------------------------------------------------------------------------
__global__ void dwconv_silu_kernel(const float* __restrict__ xz,
                                   const float* __restrict__ cw,
                                   const float* __restrict__ cb,
                                   float* __restrict__ xc) {
  long idx = (long)blockIdx.x * blockDim.x + threadIdx.x;
  if (idx >= (long)MTOT * DD) return;
  int d = (int)(idx % DD);
  long bp = idx / DD;
  int p = (int)(bp % LL);
  int b = (int)(bp / LL);
  int hi = p >> 6, wi = p & 63;
  float acc = cb[d];
#pragma unroll
  for (int dy = 0; dy < 3; dy++) {
    int hh = hi + dy - 1;
    if (hh < 0 || hh >= HH) continue;
#pragma unroll
    for (int dx = 0; dx < 3; dx++) {
      int ww = wi + dx - 1;
      if (ww < 0 || ww >= WW) continue;
      acc += cw[d * 9 + dy * 3 + dx] * xz[((long)b * LL + hh * WW + ww) * (2 * DD) + d];
    }
  }
  xc[idx] = silu_f(acc);
}

// scan index -> row-major spatial position for each direction
__device__ __forceinline__ int pos_of(int kdir, int s) {
  int t = (kdir & 2) ? (LL - 1 - s) : s;
  if (kdir & 1) {  // column-major traversal
    int hi = t & 63, wi = t >> 6;
    return hi * WW + wi;
  }
  return t;
}

// ---------------------------------------------------------------------------
// Selective scan: one lane per (b,kdir,d); N=16 state in VGPRs; dt recomputed
// per step from the rank-6 raw projection. Plain stores -> deterministic.
// ---------------------------------------------------------------------------
__global__ void scan_kernel(const float* __restrict__ xc,
                            const float* __restrict__ xdbl,
                            const float* __restrict__ dtw_all,
                            const float* __restrict__ dtb_all,
                            const float* __restrict__ A_logs,
                            const float* __restrict__ Ds,
                            float* __restrict__ yk, long ystride) {
  int wid = (int)((blockIdx.x * blockDim.x + threadIdx.x) >> 5);
  int lane = threadIdx.x & 31;
  if (wid >= BB * KDIRS * (DD / 32)) return;
  int dgrp = wid % (DD / 32);
  int kdir = (wid / (DD / 32)) & 3;
  int b = wid / (KDIRS * (DD / 32));
  int d = dgrp * 32 + lane;

  float dtw[RRANK];
#pragma unroll
  for (int r = 0; r < RRANK; r++) dtw[r] = dtw_all[(long)(kdir * DD + d) * RRANK + r];
  float bias = dtb_all[kdir * DD + d];
  float An[NSTATE], hst[NSTATE];
#pragma unroll
  for (int n = 0; n < NSTATE; n++) {
    An[n] = -__expf(A_logs[(long)(kdir * DD + d) * NSTATE + n]);
    hst[n] = 0.f;
  }
  float Dk = Ds[kdir * DD + d];

  const float* xdk = xdbl + (long)kdir * MTOT * XDBL_W + (long)b * LL * XDBL_W;
  const float* xcb = xc + (long)b * LL * DD;
  float* yb = yk + (long)kdir * ystride + (long)b * LL * DD;

  for (int s = 0; s < LL; s++) {
    int p = pos_of(kdir, s);
    if (s + 1 < LL) {  // prefetch next step's operands (global_prefetch_b8)
      int pn = pos_of(kdir, s + 1);
      __builtin_prefetch(xdk + (long)pn * XDBL_W, 0, 1);
      __builtin_prefetch(&xcb[(long)pn * DD + d], 0, 1);
    }
    const float* row = xdk + (long)p * XDBL_W;  // [6 dt_raw | 16 B | 16 C | pad]
    float u = xcb[(long)p * DD + d];
    float pre = bias;
#pragma unroll
    for (int r = 0; r < RRANK; r++) pre += dtw[r] * row[r];
    float dt = (pre > 20.f) ? pre : log1pf(__expf(pre));  // softplus
    float du = dt * u;
    float yv = 0.f;
#pragma unroll
    for (int n = 0; n < NSTATE; n++) {
      hst[n] = __expf(dt * An[n]) * hst[n] + du * row[RRANK + n];
      yv += hst[n] * row[RRANK + NSTATE + n];
    }
    yb[(long)p * DD + d] = yv + Dk * u;
  }
}

// ---------------------------------------------------------------------------
// Fused 4-direction merge + LayerNorm(192) + SiLU(z) gate; in place into yk0.
// ---------------------------------------------------------------------------
__global__ void ln_gate_kernel(float* __restrict__ yk, long ystride,
                               const float* __restrict__ xz,
                               const float* __restrict__ g,
                               const float* __restrict__ beta) {
  int wid = (int)((blockIdx.x * blockDim.x + threadIdx.x) >> 5);
  int lane = threadIdx.x & 31;
  if (wid >= MTOT) return;
  long rowoff = (long)wid * DD;
  float vals[6];
  float s = 0.f, sq = 0.f;
#pragma unroll
  for (int i = 0; i < 6; i++) {
    int c = lane + 32 * i;
    float v = yk[rowoff + c] + yk[ystride + rowoff + c] +
              yk[2 * ystride + rowoff + c] + yk[3 * ystride + rowoff + c];
    vals[i] = v;
    s += v;
    sq += v * v;
  }
  s = wred(s);
  sq = wred(sq);
  float mean = s * (1.f / DD);
  float var = sq * (1.f / DD) - mean * mean;
  float rstd = rsqrtf(var + 1e-5f);
  const float* zrow = xz + (long)wid * (2 * DD) + DD;
#pragma unroll
  for (int i = 0; i < 6; i++) {
    int c = lane + 32 * i;
    yk[rowoff + c] = ((vals[i] - mean) * rstd * g[c] + beta[c]) * silu_f(zrow[c]);
  }
}

// ---------------------------------------------------------------------------
// 2x pixel shuffle + LayerNorm(48). One wave per output pixel.
// ---------------------------------------------------------------------------
__global__ void expand_ln_kernel(const float* __restrict__ xp,
                                 const float* __restrict__ g,
                                 const float* __restrict__ beta,
                                 float* __restrict__ out) {
  int wid = (int)((blockIdx.x * blockDim.x + threadIdx.x) >> 5);
  int lane = threadIdx.x & 31;
  if (wid >= MTOT * 4) return;
  int q = wid;
  int p2 = q & 1; q >>= 1;
  int p1 = q & 1; q >>= 1;
  int w = q & 63; q >>= 6;
  int h = q & 63; q >>= 6;
  int b = q;
  const float* src = xp + ((long)(b * LL + h * WW + w)) * (2 * CC) + p1 * CC + p2 * (CC / 2);
  float v0 = src[lane];
  float v1 = (lane < 16) ? src[32 + lane] : 0.f;
  float s = wred(v0 + v1);
  float sq = wred(v0 * v0 + v1 * v1);
  float mean = s * (1.f / 48.f);
  float var = sq * (1.f / 48.f) - mean * mean;
  float rstd = rsqrtf(var + 1e-5f);
  float* orow = out + ((long)((b * 2 * HH + 2 * h + p1) * (2 * WW) + 2 * w + p2)) * (CC / 2);
  orow[lane] = (v0 - mean) * rstd * g[lane] + beta[lane];
  if (lane < 16) orow[32 + lane] = (v1 - mean) * rstd * g[32 + lane] + beta[32 + lane];
}

extern "C" void kernel_launch(void* const* d_in, const int* in_sizes, int n_in,
                              void* d_out, int out_size, void* d_ws, size_t ws_size,
                              hipStream_t stream) {
  (void)in_sizes; (void)n_in; (void)out_size; (void)ws_size;
  const float* x      = (const float*)d_in[0];   // [8,64,64,96]
  const float* W_in   = (const float*)d_in[1];   // [96,384]
  const float* conv_w = (const float*)d_in[2];   // [192,1,3,3]
  const float* conv_b = (const float*)d_in[3];   // [192]
  const float* xpw    = (const float*)d_in[4];   // [4,38,192]
  const float* dtw    = (const float*)d_in[5];   // [4,192,6]
  const float* dtb    = (const float*)d_in[6];   // [4,192]
  const float* Alogs  = (const float*)d_in[7];   // [4,192,16]
  const float* Ds     = (const float*)d_in[8];   // [4,192]
  const float* ong    = (const float*)d_in[9];   // [192]
  const float* onb    = (const float*)d_in[10];  // [192]
  const float* W_out  = (const float*)d_in[11];  // [192,96]
  const float* W_exp  = (const float*)d_in[12];  // [96,192]
  const float* eg     = (const float*)d_in[13];  // [48]
  const float* eb     = (const float*)d_in[14];  // [48]
  float* out = (float*)d_out;                    // [8,128,128,48]

  float* ws      = (float*)d_ws;
  float* xz      = ws + OFF_XZ;
  float* xc      = ws + OFF_XC;
  float* xdbl    = ws + OFF_XDBL;
  float* yk      = ws + OFF_YK;
  float* yg      = yk;                     // in-place LN+gate result
  float* out1    = yk + SZ_YK;             // reuses yk[1]
  float* xp      = xz;                     // reuses xz region
  float* W_inT   = ws + OFF_WT;
  float* W_outT  = W_inT + SZ_WIN_T;
  float* W_expT  = W_outT + SZ_WOUT_T;
  float* xpw_pad = W_expT + SZ_WEXP_T;

  // 0) weight prep: transpose to [N,K] so B-fragment loads are contiguous float4s
  transpose_kernel<<<(int)((SZ_WIN_T + 255) / 256), 256, 0, stream>>>(W_in, W_inT, CC, 2 * DD);
  transpose_kernel<<<(int)((SZ_WOUT_T + 255) / 256), 256, 0, stream>>>(W_out, W_outT, DD, CC);
  transpose_kernel<<<(int)((SZ_WEXP_T + 255) / 256), 256, 0, stream>>>(W_exp, W_expT, CC, 2 * CC);
  pad_xpw_kernel<<<(int)((SZ_XPW_P + 255) / 256), 256, 0, stream>>>(xpw, xpw_pad);

  // 1) in_proj: xz[32768,384] = x[32768,96] @ W_in (Bt = W_inT [384,96])
  {
    int tiles = (MTOT / 16) * (2 * DD / 16);  // 2048*24
    wmma_gemm_kernel<CC><<<dim3((tiles + 3) / 4, 1), 128, 0, stream>>>(
        x, W_inT, xz, MTOT / 16, 2 * DD / 16, 2 * DD, 0, 0);
  }
  // 2) depthwise 3x3 + SiLU
  dwconv_silu_kernel<<<(int)(((long)MTOT * DD + 255) / 256), 256, 0, stream>>>(
      xz, conv_w, conv_b, xc);
  // 3) x_proj per direction: xdbl[k][32768,48] = xc[32768,192] @ Wk^T (Bt = xpw_pad[k] [48,192])
  {
    int tiles = (MTOT / 16) * (XDBL_W / 16);  // 2048*3
    wmma_gemm_kernel<DD><<<dim3((tiles + 3) / 4, KDIRS), 128, 0, stream>>>(
        xc, xpw_pad, xdbl, MTOT / 16, XDBL_W / 16, XDBL_W,
        (long)XDBL_W * DD, (long)MTOT * XDBL_W);
  }
  // 4) selective scan: 192 waves, 16-state per lane in VGPRs
  scan_kernel<<<48, 128, 0, stream>>>(xc, xdbl, dtw, dtb, Alogs, Ds, yk, SZ_YK);
  // 5) merge 4 directions + LayerNorm(192) + SiLU(z) gate
  ln_gate_kernel<<<MTOT / 4, 128, 0, stream>>>(yk, SZ_YK, xz, ong, onb);
  // 6) out_proj: out1[32768,96] = yg[32768,192] @ W_out (Bt = W_outT [96,192])
  {
    int tiles = (MTOT / 16) * (CC / 16);
    wmma_gemm_kernel<DD><<<dim3((tiles + 3) / 4, 1), 128, 0, stream>>>(
        yg, W_outT, out1, MTOT / 16, CC / 16, CC, 0, 0);
  }
  // 7) expand: xp[32768,192] = out1[32768,96] @ W_exp (Bt = W_expT [192,96])
  {
    int tiles = (MTOT / 16) * (2 * CC / 16);
    wmma_gemm_kernel<CC><<<dim3((tiles + 3) / 4, 1), 128, 0, stream>>>(
        out1, W_expT, xp, MTOT / 16, 2 * CC / 16, 2 * CC, 0, 0);
  }
  // 8) pixel shuffle + LayerNorm(48) -> d_out
  expand_ln_kernel<<<MTOT, 128, 0, stream>>>(xp, eg, eb, out);
}
